// MultiHeadAttention_51445118271624
// MI455X (gfx1250) — matrix operand order repaired and checked by reading, not compiled, not run on hardware
//
#include <hip/hip_runtime.h>

// ---------------------------------------------------------------------------
// MI455X (gfx1250, wave32) attention block:
//   q=XWq, k=XWk, v=XWv ; RMS-norm(q,k) ; GQA flash-attn ; out = AO @ Wo
// All matmuls via v_wmma_f32_16x16x32_bf16; streams prefetched with
// global_prefetch_b8 (L2-resident working set, 192 MB L2, 23.3 TB/s HBM).
// ---------------------------------------------------------------------------

typedef __attribute__((ext_vector_type(16))) __bf16 v16bf;
typedef __attribute__((ext_vector_type(8)))  __bf16 v8bf;
typedef __attribute__((ext_vector_type(4)))  __bf16 v4bf;
typedef __attribute__((ext_vector_type(8)))  float  v8f;

static __device__ __forceinline__ v8f wmma_bf16(v16bf a, v16bf b, v8f c) {
  return __builtin_amdgcn_wmma_f32_16x16x32_bf16(
      /*neg_a=*/false, a, /*neg_b=*/false, b,
      /*c_mod=*/(short)0, c, /*reuse_a=*/false, /*reuse_b=*/false);
}

// A-fragment (16x32, 16-bit): lane row = lane&15;
//   elems[j] = row[kb + half*16 + j]
static __device__ __forceinline__ v16bf load_afrag(const __bf16* rowp, int kb, int half) {
  return *(const v16bf*)(rowp + kb + half * 16);
}

// B-fragment (32x16): lane col = lane&15, elems[j] = B[kb + half*16 + j][col].
// With B stored transposed (row = col, contiguous k) this is one 32B load.
static __device__ __forceinline__ v16bf load_bfrag(const __bf16* rowp, int kb, int half) {
  return *(const v16bf*)(rowp + kb + half * 16);
}

// ------------------------- problem dimensions ------------------------------
constexpr int Bb   = 2;
constexpr int Tq   = 2048;
constexpr int T    = Bb * Tq;   // 4096 tokens
constexpr int Cc   = 2048;
constexpr int Hh   = 16;
constexpr int Dd   = 128;
constexpr int HKV  = 4;
constexpr int NDQ  = Hh * Dd;   // 2048
constexpr int NDKV = HKV * Dd;  // 512

// ------------------------- fp32 -> bf16 convert ----------------------------
__global__ __launch_bounds__(256) void cvt_f32_bf16(const float* __restrict__ in,
                                                    __bf16* __restrict__ out,
                                                    size_t n4) {
  size_t i = (size_t)blockIdx.x * 256 + threadIdx.x;
  size_t stride = (size_t)gridDim.x * 256;
  for (; i < n4; i += stride) {
    float4 v = ((const float4*)in)[i];
    v4bf o;
    o[0] = (__bf16)v.x; o[1] = (__bf16)v.y; o[2] = (__bf16)v.z; o[3] = (__bf16)v.w;
    ((v4bf*)out)[i] = o;
  }
}

// ------------- fp32 (K x N) -> bf16 transposed (N x K) ---------------------
__global__ __launch_bounds__(256) void transpose_cvt(const float* __restrict__ in,
                                                     __bf16* __restrict__ out,
                                                     int K, int N) {
  __shared__ float tile[32][33];
  int tx = threadIdx.x & 31;
  int ty = threadIdx.x >> 5;                 // 8 rows of the 32x32 tile
  int n0 = blockIdx.x * 32;
  int k0 = blockIdx.y * 32;
#pragma unroll
  for (int j = 0; j < 4; ++j)
    tile[ty + j * 8][tx] = in[(size_t)(k0 + ty + j * 8) * N + n0 + tx];
  __syncthreads();
#pragma unroll
  for (int j = 0; j < 4; ++j)
    out[(size_t)(n0 + ty + j * 8) * K + k0 + tx] = (__bf16)tile[tx][ty + j * 8];
}

// ------------------------------ WMMA GEMM ----------------------------------
// C[M,N] = A[M,K](bf16,row-major) * Bt[N,K](bf16,row-major = B transposed)
// mode 0: Cf f32 row-major [M,N];  mode 1: Ct bf16 TRANSPOSED [N,M]
// Block: 256 thr = 8 waves (2m x 4n); block tile 128x128; wave tile 64x32.
__global__ __launch_bounds__(256) void gemm_wmma(const __bf16* __restrict__ A,
                                                 const __bf16* __restrict__ Bt,
                                                 float* __restrict__ Cf,
                                                 __bf16* __restrict__ Ct,
                                                 int M, int N, int K, int mode) {
  const int wave = threadIdx.x >> 5;
  const int lane = threadIdx.x & 31;
  const int col  = lane & 15;
  const int half = lane >> 4;
  const int wm = wave >> 2;          // 0..1
  const int wn = wave & 3;           // 0..3
  const int m0 = blockIdx.x * 128 + wm * 64;
  const int n0 = blockIdx.y * 128 + wn * 32;

  const __bf16* aRow[4];
  const __bf16* bRow[2];
#pragma unroll
  for (int fm = 0; fm < 4; ++fm) aRow[fm] = A + (size_t)(m0 + fm * 16 + col) * K;
#pragma unroll
  for (int fn = 0; fn < 2; ++fn) bRow[fn] = Bt + (size_t)(n0 + fn * 16 + col) * K;

  v8f acc[4][2];
#pragma unroll
  for (int fm = 0; fm < 4; ++fm)
#pragma unroll
    for (int fn = 0; fn < 2; ++fn) acc[fm][fn] = (v8f)0.0f;

  constexpr int PF = 256;  // prefetch distance: 8 K-steps (512 B) ahead
  for (int kb = 0; kb < K; kb += 32) {
    if (kb + PF < K) {
#pragma unroll
      for (int fm = 0; fm < 4; ++fm) __builtin_prefetch(aRow[fm] + kb + PF, 0, 1);
#pragma unroll
      for (int fn = 0; fn < 2; ++fn) __builtin_prefetch(bRow[fn] + kb + PF, 0, 1);
    }
    v16bf bf[2];
#pragma unroll
    for (int fn = 0; fn < 2; ++fn) bf[fn] = load_bfrag(bRow[fn], kb, half);
    v16bf af[4];
#pragma unroll
    for (int fm = 0; fm < 4; ++fm) af[fm] = load_afrag(aRow[fm], kb, half);
#pragma unroll
    for (int fm = 0; fm < 4; ++fm)
#pragma unroll
      for (int fn = 0; fn < 2; ++fn)
        acc[fm][fn] = wmma_bf16(af[fm], bf[fn], acc[fm][fn]);
  }

  if (mode == 0) {
#pragma unroll
    for (int fm = 0; fm < 4; ++fm)
#pragma unroll
      for (int fn = 0; fn < 2; ++fn) {
        int n = n0 + fn * 16 + col;
#pragma unroll
        for (int r = 0; r < 8; ++r) {
          int m = m0 + fm * 16 + r + 8 * half;
          Cf[(size_t)m * N + n] = acc[fm][fn][r];
        }
      }
  } else {
    // transposed bf16 store: lane holds 8 consecutive M for fixed N
#pragma unroll
    for (int fm = 0; fm < 4; ++fm)
#pragma unroll
      for (int fn = 0; fn < 2; ++fn) {
        int n = n0 + fn * 16 + col;
        int mstart = m0 + fm * 16 + 8 * half;
        union { v8bf v; __bf16 e[8]; } o;
#pragma unroll
        for (int r = 0; r < 8; ++r) o.e[r] = (__bf16)acc[fm][fn][r];
        *(v8bf*)(Ct + (size_t)n * M + mstart) = o.v;
      }
  }
}

// --------------------- RMS-norm over D=128, f32 -> bf16 --------------------
__global__ __launch_bounds__(256) void rmsnorm128(const float* __restrict__ in,
                                                  __bf16* __restrict__ out,
                                                  int rows) {
  int wave = threadIdx.x >> 5;
  int lane = threadIdx.x & 31;
  int row = blockIdx.x * 8 + wave;
  if (row >= rows) return;
  const float4 v = *(const float4*)(in + (size_t)row * 128 + lane * 4);
  float ss = v.x * v.x + v.y * v.y + v.z * v.z + v.w * v.w;
#pragma unroll
  for (int off = 16; off >= 1; off >>= 1) ss += __shfl_xor(ss, off, 32);
  float s = rsqrtf(ss * (1.0f / 128.0f) + 1e-6f);
  v4bf o;
  o[0] = (__bf16)(v.x * s); o[1] = (__bf16)(v.y * s);
  o[2] = (__bf16)(v.z * s); o[3] = (__bf16)(v.w * s);
  *(v4bf*)(out + (size_t)row * 128 + lane * 4) = o;
}

// ------------------------------ flash attention ----------------------------
// qn: [T, H*D] bf16 (normalized)   kn: [T, HKV*D] bf16 (normalized)
// vt: [HKV*D, T] bf16 (V transposed)   AO: [T, H*D] bf16
// grid.x = B*H, grid.y = Tq/64 ; block = 128 thr = 4 waves, 16 q-rows/wave.
// Computes S^T = K * Q^T (both operands contiguous), online softmax over rows
// (keys) per lane column (query), P^T built with shfl_xor(16), O^T = V^T*P^T.
__global__ __launch_bounds__(128) void flash_attn(const __bf16* __restrict__ qn,
                                                  const __bf16* __restrict__ kn,
                                                  const __bf16* __restrict__ vt,
                                                  __bf16* __restrict__ AO) {
  const int wave = threadIdx.x >> 5;
  const int lane = threadIdx.x & 31;
  const int col  = lane & 15;   // query index within the 16-wide tile
  const int half = lane >> 4;
  const int b   = blockIdx.x >> 4;
  const int h   = blockIdx.x & 15;
  const int hkv = h >> 2;
  const int qtok = b * Tq + blockIdx.y * 64 + wave * 16 + col;

  const __bf16* qrow = qn + (size_t)qtok * NDQ + h * Dd;
  v16bf qf[4];
#pragma unroll
  for (int s = 0; s < 4; ++s)
    qf[s] = *(const v16bf*)(qrow + s * 32 + half * 16);   // B-frag of Q^T

  v8f oacc[8];
#pragma unroll
  for (int dt = 0; dt < 8; ++dt) oacc[dt] = (v8f)0.0f;
  float m2 = -3.0e38f;
  float l  = 0.0f;
  const float sc2 = 0.08838834764831845f * 1.4426950408889634f; // (1/sqrt(D))*log2(e)

  for (int kb0 = 0; kb0 < Tq; kb0 += 32) {
    // --- S^T = K(16x128) * Q^T, two 16-key subtiles ---
    const __bf16* kr0 = kn + (size_t)(b * Tq + kb0 + col) * NDKV + hkv * Dd;
    const __bf16* kr1 = kr0 + (size_t)16 * NDKV;
    if (kb0 + 32 < Tq) {  // prefetch next chunk's K rows + V^T stream
      __builtin_prefetch(kr0 + (size_t)32 * NDKV, 0, 1);
      __builtin_prefetch(kr1 + (size_t)32 * NDKV, 0, 1);
      __builtin_prefetch(vt + (size_t)(hkv * Dd + col) * T + b * Tq + kb0 + 32, 0, 1);
    }
    v8f st0 = (v8f)0.0f, st1 = (v8f)0.0f;
#pragma unroll
    for (int s = 0; s < 4; ++s) st0 = wmma_bf16(load_afrag(kr0, s * 32, half), qf[s], st0);
#pragma unroll
    for (int s = 0; s < 4; ++s) st1 = wmma_bf16(load_afrag(kr1, s * 32, half), qf[s], st1);

    // --- online softmax (per lane column = per query) ---
    float cm = -3.0e38f;
#pragma unroll
    for (int r = 0; r < 8; ++r) cm = fmaxf(cm, fmaxf(st0[r], st1[r]));
    cm = fmaxf(cm, __shfl_xor(cm, 16, 32));
    float m2new = fmaxf(m2, cm * sc2);
    float alpha = __builtin_amdgcn_exp2f(m2 - m2new);
    float pa[8], pb[8], sum = 0.0f;
#pragma unroll
    for (int r = 0; r < 8; ++r) {
      pa[r] = __builtin_amdgcn_exp2f(st0[r] * sc2 - m2new);
      pb[r] = __builtin_amdgcn_exp2f(st1[r] * sc2 - m2new);
      sum += pa[r] + pb[r];
    }
    sum += __shfl_xor(sum, 16, 32);
    l = l * alpha + sum;
    m2 = m2new;
#pragma unroll
    for (int dt = 0; dt < 8; ++dt)
#pragma unroll
      for (int r = 0; r < 8; ++r) oacc[dt][r] *= alpha;

    // --- build P^T B-fragment via cross-half exchange ---
    union { v16bf v; __bf16 e[16]; } pf;
#pragma unroll
    for (int r = 0; r < 8; ++r) {
      float xa = __shfl_xor(pa[r], 16, 32);
      float xb = __shfl_xor(pb[r], 16, 32);
      float lo = half ? xb : pa[r];   // keys half*16 + r
      float hv = half ? pb[r] : xa;   // keys half*16 + 8 + r
      pf.e[r]     = (__bf16)lo;
      pf.e[r + 8] = (__bf16)hv;
    }

    // --- O^T += V^T * P^T over 8 d-tiles ---
#pragma unroll
    for (int dt = 0; dt < 8; ++dt) {
      const __bf16* vrow = vt + (size_t)(hkv * Dd + dt * 16 + col) * T + b * Tq + kb0;
      oacc[dt] = wmma_bf16(load_afrag(vrow, 0, half), pf.v, oacc[dt]);
    }
  }

  const float inv = 1.0f / l;
  __bf16* orow = AO + (size_t)qtok * NDQ + h * Dd;
#pragma unroll
  for (int dt = 0; dt < 8; ++dt) {
    union { v8bf v; __bf16 e[8]; } o;
#pragma unroll
    for (int r = 0; r < 8; ++r) o.e[r] = (__bf16)(oacc[dt][r] * inv);
    *(v8bf*)(orow + dt * 16 + 8 * half) = o.v;
  }
}

// ------------------------------- launcher ----------------------------------
extern "C" void kernel_launch(void* const* d_in, const int* in_sizes, int n_in,
                              void* d_out, int out_size, void* d_ws, size_t ws_size,
                              hipStream_t stream) {
  const float* x  = (const float*)d_in[0];   // (B,Tq,C)
  const float* Wq = (const float*)d_in[1];   // (C, H*D)
  const float* Wk = (const float*)d_in[2];   // (C, HKV*D)
  const float* Wv = (const float*)d_in[3];   // (C, HKV*D)
  const float* Wo = (const float*)d_in[4];   // (H*D, C)
  float* out = (float*)d_out;

  char* ws = (char*)d_ws;
  size_t off = 0;
  auto alloc = [&](size_t bytes) { char* p = ws + off; off += (bytes + 255) & ~(size_t)255; return p; };
  __bf16* Xb   = (__bf16*)alloc((size_t)T * Cc * 2);
  __bf16* Wqt  = (__bf16*)alloc((size_t)NDQ * Cc * 2);
  __bf16* Wkt  = (__bf16*)alloc((size_t)NDKV * Cc * 2);
  __bf16* Wvt  = (__bf16*)alloc((size_t)NDKV * Cc * 2);
  __bf16* Wot  = (__bf16*)alloc((size_t)Cc * NDQ * 2);
  float*  qf32 = (float*) alloc((size_t)T * NDQ * 4);
  float*  kf32 = (float*) alloc((size_t)T * NDKV * 4);
  __bf16* qn   = (__bf16*)alloc((size_t)T * NDQ * 2);
  __bf16* kn   = (__bf16*)alloc((size_t)T * NDKV * 2);
  __bf16* vt   = (__bf16*)alloc((size_t)NDKV * T * 2);
  __bf16* AO   = (__bf16*)alloc((size_t)T * NDQ * 2);

  // 1) activations -> bf16
  cvt_f32_bf16<<<8192, 256, 0, stream>>>(x, Xb, (size_t)T * Cc / 4);
  // 2) weights -> bf16 transposed (N x K)
  transpose_cvt<<<dim3(NDQ / 32, Cc / 32), 256, 0, stream>>>(Wq, Wqt, Cc, NDQ);
  transpose_cvt<<<dim3(NDKV / 32, Cc / 32), 256, 0, stream>>>(Wk, Wkt, Cc, NDKV);
  transpose_cvt<<<dim3(NDKV / 32, Cc / 32), 256, 0, stream>>>(Wv, Wvt, Cc, NDKV);
  transpose_cvt<<<dim3(Cc / 32, NDQ / 32), 256, 0, stream>>>(Wo, Wot, NDQ, Cc);
  // 3) projections
  gemm_wmma<<<dim3(T / 128, NDQ / 128), 256, 0, stream>>>(Xb, Wqt, qf32, nullptr, T, NDQ, Cc, 0);
  gemm_wmma<<<dim3(T / 128, NDKV / 128), 256, 0, stream>>>(Xb, Wkt, kf32, nullptr, T, NDKV, Cc, 0);
  gemm_wmma<<<dim3(T / 128, NDKV / 128), 256, 0, stream>>>(Xb, Wvt, nullptr, vt, T, NDKV, Cc, 1);
  // 4) QK RMS-norm -> bf16
  rmsnorm128<<<(T * Hh) / 8, 256, 0, stream>>>(qf32, qn, T * Hh);
  rmsnorm128<<<(T * HKV) / 8, 256, 0, stream>>>(kf32, kn, T * HKV);
  // 5) GQA flash attention
  flash_attn<<<dim3(Bb * Hh, Tq / 64), 128, 0, stream>>>(qn, kn, vt, AO);
  // 6) output projection -> f32 d_out
  gemm_wmma<<<dim3(T / 128, Cc / 128), 256, 0, stream>>>(AO, Wot, out, nullptr, T, Cc, Cc, 0);

  (void)in_sizes; (void)n_in; (void)out_size; (void)ws_size;
}
